// PraxisSMEAR_85822036509273
// MI455X (gfx1250) — matrix operand order repaired and checked by reading, not compiled, not run on hardware
//
#include <hip/hip_runtime.h>
#include <math.h>

typedef __attribute__((ext_vector_type(16))) _Float16 v16h;
typedef __attribute__((ext_vector_type(8)))  _Float16 v8h;
typedef __attribute__((ext_vector_type(8)))  float    v8f;

#define D_ 1024
#define E_ 8
#define B_ 4
#define S_ 4096
#define M_ (B_ * S_)      // 16384 rows
#define CH 64             // S-chunks for pooling
#define SC (S_ / CH)      // 64 rows per chunk

// Workspace layout (float offsets)
#define WS_PARTIAL 0
#define WS_POOLED  (B_ * CH * D_)             // 262144
#define WS_W       (WS_POOLED + B_ * D_)      // 266240
#define WS_MBIAS   (WS_W + 64)                // 266304
#define WS_WHI     (WS_MBIAS + D_)            // 267328 (16B aligned)
#define WS_WLO     (WS_WHI + (D_ * D_ / 2))   // + 524288 floats

// ---- CDNA5 async global->LDS path (guarded; fallback = load + ds_store) ----
#if defined(__has_builtin)
#if __has_builtin(__builtin_amdgcn_global_load_async_to_lds_b128)
#define HAVE_ASYNC_LDS 1
#endif
#if __has_builtin(__builtin_amdgcn_s_wait_asynccnt)
#define WAIT_ASYNC0() __builtin_amdgcn_s_wait_asynccnt(0)
#endif
#endif
#if defined(HAVE_ASYNC_LDS) && !defined(WAIT_ASYNC0)
#define WAIT_ASYNC0() asm volatile("s_wait_asynccnt 0x0" ::: "memory")
#endif

#ifdef HAVE_ASYNC_LDS
typedef int v4i_ __attribute__((vector_size(16)));
typedef __attribute__((address_space(1))) v4i_* as1_v4i_ptr;
typedef __attribute__((address_space(3))) v4i_* as3_v4i_ptr;
__device__ __forceinline__ void async_copy16(const _Float16* g, _Float16* l) {
    __builtin_amdgcn_global_load_async_to_lds_b128(
        (as1_v4i_ptr)g, (as3_v4i_ptr)l, 0, 0);
}
#endif

// ---------------- Stage 1: partial sums over S chunks ----------------
__global__ __launch_bounds__(256)
void pool_partial_kernel(const float* __restrict__ in, float* __restrict__ partial) {
    const int d = blockIdx.x * 256 + threadIdx.x;
    const int c = blockIdx.y;
    const int b = blockIdx.z;
    const float* p = in + ((size_t)b * S_ + (size_t)c * SC) * D_ + d;
    float acc = 0.f;
    for (int s = 0; s < SC; ++s) acc += p[(size_t)s * D_];
    partial[((size_t)b * CH + c) * D_ + d] = acc;
}

// ---------------- Stage 2: reduce chunks -> pooled mean ----------------
__global__ __launch_bounds__(256)
void pool_reduce_kernel(const float* __restrict__ partial, float* __restrict__ pooled) {
    const int idx = blockIdx.x * 256 + threadIdx.x;   // [0, B_*D_)
    const int b = idx / D_, d = idx % D_;
    float acc = 0.f;
    for (int c = 0; c < CH; ++c) acc += partial[((size_t)b * CH + c) * D_ + d];
    pooled[idx] = acc * (1.f / (float)S_);
}

// ------- Stage 3: router logits + softmax + mixture w[E] + merged bias -------
__global__ __launch_bounds__(1024)
void router_kernel(const float* __restrict__ pooled, const float* __restrict__ rw,
                   const float* __restrict__ rb, const float* __restrict__ eb,
                   float* __restrict__ wE, float* __restrict__ mbias) {
    __shared__ float slog[B_ * E_];
    __shared__ float sw[E_];
    const int tid = threadIdx.x;
    const int L = tid & 31, wv = tid >> 5;    // 32 waves: one per (b,e)
    const int b = wv >> 3, e = wv & 7;
    float acc = 0.f;
    for (int d = L; d < D_; d += 32)
        acc += pooled[b * D_ + d] * rw[e * D_ + d];
#pragma unroll
    for (int off = 16; off; off >>= 1) acc += __shfl_xor(acc, off, 32);
    if (L == 0) slog[wv] = acc + rb[e];
    __syncthreads();
    if (tid == 0) {
        float wacc[E_];
        for (int i = 0; i < E_; ++i) wacc[i] = 0.f;
        for (int bb = 0; bb < B_; ++bb) {
            float mx = -1e30f;
            for (int ee = 0; ee < E_; ++ee) mx = fmaxf(mx, slog[bb * E_ + ee]);
            float s = 0.f, ex[E_];
            for (int ee = 0; ee < E_; ++ee) { ex[ee] = __expf(slog[bb * E_ + ee] - mx); s += ex[ee]; }
            const float inv = 1.f / s;
            for (int ee = 0; ee < E_; ++ee) wacc[ee] += ex[ee] * inv;
        }
        for (int ee = 0; ee < E_; ++ee) { float v = wacc[ee] * (1.f / (float)B_); sw[ee] = v; wE[ee] = v; }
    }
    __syncthreads();
    {   // merged bias: mbias[o] = sum_e w[e] * expert_b[e][o]  (blockDim == D_)
        float a = 0.f;
#pragma unroll
        for (int ee = 0; ee < E_; ++ee) a = fmaf(sw[ee], eb[ee * D_ + tid], a);
        mbias[tid] = a;
    }
}

// ------- Stage 4: merge experts -> split-f16 (hi/lo) merged weight matrix -------
__global__ __launch_bounds__(256)
void merge_kernel(const float* __restrict__ ew, const float* __restrict__ wE,
                  _Float16* __restrict__ whi, _Float16* __restrict__ wlo) {
    const int idx = blockIdx.x * 256 + threadIdx.x;   // [0, D_*D_)
    float acc = 0.f;
#pragma unroll
    for (int e = 0; e < E_; ++e)
        acc = fmaf(wE[e], ew[(size_t)e * (D_ * D_) + idx], acc);
    const _Float16 h = (_Float16)acc;
    whi[idx] = h;
    wlo[idx] = (_Float16)(acc - (float)h);
}

// ------- Stage 5: GEMM out[m][n] = sum_k A[m][k]*Wm[n][k] + bias[n] -------
// Block = 128x128 output tile: 8 waves, wave wv owns M-tile rows [bx*128+wv*16, +16);
// all waves share a 128-column B panel (hi+lo) double-buffered in LDS via async copies.
// Split-f16: D += Ah*Bh + Al*Bh + Ah*Bl  (3x v_wmma_f32_16x16x32_f16 per tile per K-step).
#define ROWH   40                    // halfs per LDS row (32 data + 8 pad, bank spread)
#define ARRH   (128 * ROWH)          // halfs per (hi|lo) array = 5120
#define BUFH   (2 * ARRH)            // halfs per stage buffer (hi + lo) = 10240
#define NSTAGE (D_ / 32)             // 32 K-steps

__global__ __launch_bounds__(256)
void gemm_kernel(const float* __restrict__ A, const _Float16* __restrict__ whi,
                 const _Float16* __restrict__ wlo, const float* __restrict__ mbias,
                 float* __restrict__ out) {
    __shared__ _Float16 lds[2 * BUFH];   // 2 stages x (hi+lo) x 128 rows x 40 halfs = 40KB

    const int tid  = threadIdx.x;
    const int L    = tid & 31;
    const int wv   = tid >> 5;
    const int lmod = L & 15;
    const int lhi  = L >> 4;             // lane half selects K sub-range
    const int m0w  = blockIdx.x * 128 + wv * 16;
    const int n0   = blockIdx.y * 128;
    const int koff = lhi * 8;            // A: lanes0-15 -> K {0..7,16..23}; 16-31 -> {8..15,24..31}
    const float* arow = A + (size_t)(m0w + lmod) * D_;

    // Staging role: thread tid copies one 32-half half-row of the B panel (hi and lo).
    const int srow  = tid >> 1;          // 0..127  (panel row = Wm row n0+srow)
    const int shalf = tid & 1;           // 0..1    (k sub-halves 0..15 / 16..31)
    const _Float16* ghi_base = whi + (size_t)(n0 + srow) * D_ + shalf * 16;
    const _Float16* glo_base = wlo + (size_t)(n0 + srow) * D_ + shalf * 16;
    const int lofs = srow * ROWH + shalf * 16;   // half offset within an array

    v8f c[8];
#pragma unroll
    for (int t = 0; t < 8; ++t) c[t] = {};

#ifdef HAVE_ASYNC_LDS
#define STAGE_ISSUE(ki, bi)                                                  \
    do {                                                                     \
        const _Float16* gh = ghi_base + (ki) * 32;                           \
        const _Float16* gl = glo_base + (ki) * 32;                           \
        _Float16* lh = lds + (bi) * BUFH + lofs;                             \
        _Float16* ll = lh + ARRH;                                            \
        async_copy16(gh, lh);  async_copy16(gh + 8, lh + 8);                 \
        async_copy16(gl, ll);  async_copy16(gl + 8, ll + 8);                 \
    } while (0)
#define STAGE_WAIT() WAIT_ASYNC0()
#else
#define STAGE_ISSUE(ki, bi)                                                  \
    do {                                                                     \
        const v8h h0 = *(const v8h*)(ghi_base + (ki) * 32);                  \
        const v8h h1 = *(const v8h*)(ghi_base + (ki) * 32 + 8);              \
        const v8h l0 = *(const v8h*)(glo_base + (ki) * 32);                  \
        const v8h l1 = *(const v8h*)(glo_base + (ki) * 32 + 8);              \
        _Float16* lh = lds + (bi) * BUFH + lofs;                             \
        _Float16* ll = lh + ARRH;                                            \
        *(v8h*)lh = h0;  *(v8h*)(lh + 8) = h1;                               \
        *(v8h*)ll = l0;  *(v8h*)(ll + 8) = l1;                               \
    } while (0)
#define STAGE_WAIT() do {} while (0)
#endif

    STAGE_ISSUE(0, 0);

    for (int i = 0; i < NSTAGE; ++i) {
        STAGE_WAIT();           // this wave's stage-i async copies have landed in LDS
        __syncthreads();        // all waves' stage-i data visible; all done reading buf[(i+1)&1]
        if (i + 1 < NSTAGE) STAGE_ISSUE(i + 1, (i + 1) & 1);

        // ---- A fragment for kk = i*32: 16 f32 per lane, split into f16 hi/lo ----
        const int kk = i * 32;
        float av[16];
        {
            const float4 a0 = *(const float4*)(arow + kk + koff);
            const float4 a1 = *(const float4*)(arow + kk + koff + 4);
            const float4 a2 = *(const float4*)(arow + kk + 16 + koff);
            const float4 a3 = *(const float4*)(arow + kk + 16 + koff + 4);
            av[0]=a0.x;  av[1]=a0.y;  av[2]=a0.z;  av[3]=a0.w;
            av[4]=a1.x;  av[5]=a1.y;  av[6]=a1.z;  av[7]=a1.w;
            av[8]=a2.x;  av[9]=a2.y;  av[10]=a2.z; av[11]=a2.w;
            av[12]=a3.x; av[13]=a3.y; av[14]=a3.z; av[15]=a3.w;
        }
        v16h ah, al;
#pragma unroll
        for (int j = 0; j < 16; ++j) {
            const _Float16 h = (_Float16)av[j];
            ah[j] = h;
            al[j] = (_Float16)(av[j] - (float)h);
        }
        if (kk + 32 < D_) __builtin_prefetch(arow + kk + 32 + koff, 0, 1);

        // ---- 8 N-tiles from the LDS panel ----
        const _Float16* lb = lds + (i & 1) * BUFH;
#pragma unroll
        for (int t = 0; t < 8; ++t) {
            const _Float16* bp  = lb + (t * 16 + lmod) * ROWH + lhi * 8;    // hi
            const _Float16* bpl = bp + ARRH;                                 // lo
            const v8h h0 = *(const v8h*)bp,  h1 = *(const v8h*)(bp + 16);
            const v8h l0 = *(const v8h*)bpl, l1 = *(const v8h*)(bpl + 16);
            const v16h bh = __builtin_shufflevector(h0, h1, 0,1,2,3,4,5,6,7,8,9,10,11,12,13,14,15);
            const v16h bl = __builtin_shufflevector(l0, l1, 0,1,2,3,4,5,6,7,8,9,10,11,12,13,14,15);
            c[t] = __builtin_amdgcn_wmma_f32_16x16x32_f16(false, ah, false, bh, (short)0, c[t], false, false);
            c[t] = __builtin_amdgcn_wmma_f32_16x16x32_f16(false, al, false, bh, (short)0, c[t], false, false);
            c[t] = __builtin_amdgcn_wmma_f32_16x16x32_f16(false, ah, false, bl, (short)0, c[t], false, false);
        }
    }

    // ---- Epilogue: C layout lanes0-15 N=lane M=v, lanes16-31 N=lane-16 M=8+v ----
#pragma unroll
    for (int t = 0; t < 8; ++t) {
        const int n  = n0 + t * 16 + lmod;
        const float bb = mbias[n];
#pragma unroll
        for (int v = 0; v < 8; ++v) {
            const int m = m0w + lhi * 8 + v;
            out[(size_t)m * D_ + n] = c[t][v] + bb;
        }
    }
}

extern "C" void kernel_launch(void* const* d_in, const int* in_sizes, int n_in,
                              void* d_out, int out_size, void* d_ws, size_t ws_size,
                              hipStream_t stream) {
    const float* inputs   = (const float*)d_in[0];
    const float* router_w = (const float*)d_in[1];
    const float* router_b = (const float*)d_in[2];
    const float* expert_w = (const float*)d_in[3];
    const float* expert_b = (const float*)d_in[4];
    float* out = (float*)d_out;

    float* wsf     = (float*)d_ws;
    float* partial = wsf + WS_PARTIAL;
    float* pooled  = wsf + WS_POOLED;
    float* wE      = wsf + WS_W;
    float* mbias   = wsf + WS_MBIAS;
    _Float16* whi  = (_Float16*)(wsf + WS_WHI);
    _Float16* wlo  = (_Float16*)(wsf + WS_WLO);

    pool_partial_kernel<<<dim3(D_ / 256, CH, B_), 256, 0, stream>>>(inputs, partial);
    pool_reduce_kernel<<<(B_ * D_) / 256, 256, 0, stream>>>(partial, pooled);
    router_kernel<<<1, 1024, 0, stream>>>(pooled, router_w, router_b, expert_b, wE, mbias);
    merge_kernel<<<(D_ * D_) / 256, 256, 0, stream>>>(expert_w, wE, whi, wlo);
    gemm_kernel<<<dim3(M_ / 128, D_ / 128), 256, 0, stream>>>(inputs, whi, wlo, mbias, out);
}
